// SelfAttention_1297080123519
// MI455X (gfx1250) — matrix operand order repaired and checked by reading, not compiled, not run on hardware
//
#include <hip/hip_runtime.h>
#include <stdint.h>

// ---------------------------------------------------------------------------
// Problem constants: B=8, C=192, H=W=128, heads=4, hd=48
// ---------------------------------------------------------------------------
#define B_DIM 8
#define C_DIM 192
#define N_PIX 16384          // 128*128
#define OC3   576            // 3*C
#define HEADS 4
#define HD    48
#define KCHUNK 2048          // QK^T split-K chunk
#define NCHUNK (N_PIX / KCHUNK)

typedef __bf16 bf16_t;
typedef __attribute__((ext_vector_type(16))) __bf16 v16bf;
typedef __attribute__((ext_vector_type(8)))  __bf16 v8bf;
typedef __attribute__((ext_vector_type(8)))  float  v8f;
typedef __attribute__((ext_vector_type(4)))  unsigned int u32x4;
typedef __attribute__((ext_vector_type(8)))  int i32x8;
typedef __attribute__((ext_vector_type(4)))  int i32x4;

// Native converts: gfx1250 has hardware bf16 cvt; let the backend pick it.
static __device__ __forceinline__ bf16_t f2bf(float f) { return (bf16_t)f; }
static __device__ __forceinline__ float bf2f(bf16_t b) { return (float)b; }

static __device__ __forceinline__ v8f wmma_bf16(v16bf a, v16bf b, v8f c) {
  return __builtin_amdgcn_wmma_f32_16x16x32_bf16(false, a, false, b, (short)0, c,
                                                 false, false);
}

// Wave32 fragment layouts used here (contiguous k order, same in A and B):
//  A 16x32 bf16 : lane l -> m=l&15 ; elem i -> k = (l>>4)*16 + i
//    => per lane: one contiguous 16-element run at k0+(half*16)
//  B 32x16 bf16 : lane l -> n=l&15 ; elem i -> k = (l>>4)*16 + i
//    => per lane: one contiguous 16-element run at k0+(half*16)
//  D 16x16 f32  : elem r -> m = r + (l>>4)*8 ; n = l&15

// A fragment from a contiguous bf16 row (two 16B vector loads)
static __device__ __forceinline__ v16bf load_a_bf16(const bf16_t* row, int k0, int half) {
  const v8bf* p = (const v8bf*)(row + k0 + (half << 4));
  v8bf lo = p[0];         // k0+half*16 .. +7
  v8bf hi = p[1];         // +8
  return __builtin_shufflevector(lo, hi, 0, 1, 2, 3, 4, 5, 6, 7,
                                 8, 9, 10, 11, 12, 13, 14, 15);
}
// B fragment from a contiguous bf16 row (one 32B vector load)
static __device__ __forceinline__ v16bf load_b_bf16(const bf16_t* row, int k0, int half) {
  return *(const v16bf*)(row + k0 + (half << 4));
}
// A fragment from a contiguous f32 row (four float4 loads + convert)
static __device__ __forceinline__ v16bf load_a_f32(const float* row, int k0, int half) {
  const float4* p = (const float4*)(row + k0 + (half << 4));
  float4 x0 = p[0], x1 = p[1], x2 = p[2], x3 = p[3];
  v16bf a;
  a[0] = f2bf(x0.x); a[1] = f2bf(x0.y); a[2]  = f2bf(x0.z); a[3]  = f2bf(x0.w);
  a[4] = f2bf(x1.x); a[5] = f2bf(x1.y); a[6]  = f2bf(x1.z); a[7]  = f2bf(x1.w);
  a[8] = f2bf(x2.x); a[9] = f2bf(x2.y); a[10] = f2bf(x2.z); a[11] = f2bf(x2.w);
  a[12] = f2bf(x3.x); a[13] = f2bf(x3.y); a[14] = f2bf(x3.z); a[15] = f2bf(x3.w);
  return a;
}

// ---------------------------------------------------------------------------
// K1: qkv[b,o,p] = sum_c w_qkv[o,c] * x[b,c,p]   (576x16384, K=192), bf16 out
// grid (1024, 9, 8), block 128 (4 waves, one 16x16 tile each)
// ---------------------------------------------------------------------------
__global__ void qkv_gemm_kernel(const float* __restrict__ x,
                                const float* __restrict__ w,
                                bf16_t* __restrict__ qkv) {
  const int lane = threadIdx.x & 31, wave = threadIdx.x >> 5;
  const int half = lane >> 4, ln = lane & 15;
  const int n0 = blockIdx.x * 16;
  const int m0 = (blockIdx.y * 4 + wave) * 16;
  const int b  = blockIdx.z;
  const float* wrow = w + (size_t)(m0 + ln) * C_DIM;
  const float* xb   = x + (size_t)b * C_DIM * N_PIX + n0 + ln;

  v8f acc = {};
  for (int k0 = 0; k0 < C_DIM; k0 += 32) {
    v16bf a = load_a_f32(wrow, k0, half);
    v16bf bb;
#pragma unroll
    for (int i = 0; i < 16; ++i) {   // strided across k; coalesced across lanes
      const int kb = k0 + (half << 4) + i;
      bb[i] = f2bf(xb[(size_t)kb * N_PIX]);
    }
    acc = wmma_bf16(a, bb, acc);
  }
  bf16_t* op = qkv + (size_t)b * OC3 * N_PIX;
#pragma unroll
  for (int r = 0; r < 8; ++r)
    op[(size_t)(m0 + r + (half << 3)) * N_PIX + n0 + ln] = f2bf(acc[r]);
}

// ---------------------------------------------------------------------------
// K2: 3x3 depthwise conv, SAME padding. bf16 in/out, f32 accumulate.
// ---------------------------------------------------------------------------
__global__ void dwconv_kernel(const bf16_t* __restrict__ qkv,
                              const float* __restrict__ wdw,
                              bf16_t* __restrict__ out) {
  const size_t idx = (size_t)blockIdx.x * blockDim.x + threadIdx.x;
  const int xw = (int)(idx & 127);
  const int yh = (int)((idx >> 7) & 127);
  const int t  = (int)(idx >> 14);
  const int ch = t % OC3;
  const int b  = t / OC3;
  const bf16_t* in = qkv + ((size_t)(b * OC3 + ch) << 14);
  const float*  wp = wdw + ch * 9;
  float s = 0.f;
#pragma unroll
  for (int dy = -1; dy <= 1; ++dy) {
#pragma unroll
    for (int dx = -1; dx <= 1; ++dx) {
      const int yy = yh + dy, xx = xw + dx;
      if (yy >= 0 && yy < 128 && xx >= 0 && xx < 128)
        s += wp[(dy + 1) * 3 + (dx + 1)] * bf2f(in[(yy << 7) + xx]);
    }
  }
  out[idx] = f2bf(s);
}

// ---------------------------------------------------------------------------
// K3: inverse L2 norms of q/k rows. id = sel*1536 + b*192 + ch.
// ---------------------------------------------------------------------------
__global__ void rownorm_kernel(const bf16_t* __restrict__ qkv2,
                               float* __restrict__ inv) {
  const int id  = blockIdx.x;
  const int ch  = id % C_DIM;
  const int t   = id / C_DIM;
  const int b   = t & 7;
  const int sel = t >> 3;
  const bf16_t* row = qkv2 + ((size_t)(b * OC3 + sel * C_DIM + ch) << 14);
  float s = 0.f;
  for (int i = threadIdx.x * 8; i < N_PIX; i += 256 * 8) {
    v8bf v = *(const v8bf*)(row + i);
#pragma unroll
    for (int j = 0; j < 8; ++j) {
      const float f = bf2f(v[j]);
      s += f * f;
    }
  }
  __shared__ float red[256];
  red[threadIdx.x] = s;
  __syncthreads();
  for (int off = 128; off > 0; off >>= 1) {
    if ((int)threadIdx.x < off) red[threadIdx.x] += red[threadIdx.x + off];
    __syncthreads();
  }
  if (threadIdx.x == 0) inv[id] = 1.f / fmaxf(sqrtf(red[0]), 1e-12f);
}

// ---------------------------------------------------------------------------
// K4a: partial QK^T logits, split-K. grid (8 chunks, 32 bh), block 288.
// Vectorized v16bf fragment loads; f32 atomics into logits[bh][48][48].
// ---------------------------------------------------------------------------
__global__ void attn_partial_kernel(const bf16_t* __restrict__ qkv2,
                                    float* __restrict__ logits) {
  const int chunk = blockIdx.x;
  const int bh = blockIdx.y, b = bh >> 2, h = bh & 3;
  const int lane = threadIdx.x & 31, wave = threadIdx.x >> 5;
  const int m0 = (wave / 3) * 16, n0 = (wave % 3) * 16;
  const int half = lane >> 4, ln = lane & 15;
  const bf16_t* qrow = qkv2 + ((size_t)(b * OC3 + h * HD + m0 + ln) << 14);
  const bf16_t* krow = qkv2 + ((size_t)(b * OC3 + C_DIM + h * HD + n0 + ln) << 14);
  const int kbeg = chunk * KCHUNK, kend = kbeg + KCHUNK;

  v8f acc = {};
#pragma unroll 2
  for (int k0 = kbeg; k0 < kend; k0 += 32) {
    __builtin_prefetch(qrow + k0 + 1024, 0, 1);    // global_prefetch_b8
    __builtin_prefetch(krow + k0 + 1024, 0, 1);
    v16bf a  = load_a_bf16(qrow, k0, half);
    v16bf bb = load_b_bf16(krow, k0, half);
    acc = wmma_bf16(a, bb, acc);
  }
  float* lp = logits + (size_t)bh * 48 * 48;
#pragma unroll
  for (int r = 0; r < 8; ++r) {
    const int c = m0 + r + (half << 3);
    const int d = n0 + ln;
    unsafeAtomicAdd(lp + c * 48 + d, acc[r]);      // global_atomic_add_f32
  }
}

// ---------------------------------------------------------------------------
// K4b: scale by invq*invk*temp and softmax over d; write attn to d_out tail.
// grid 32, block 64 (rows 0..47 active).
// ---------------------------------------------------------------------------
__global__ void attn_softmax_kernel(const float* __restrict__ logits,
                                    const float* __restrict__ inv,
                                    const float* __restrict__ temp,
                                    float* __restrict__ attn_out) {
  const int bh = blockIdx.x, b = bh >> 2, h = bh & 3;
  if (threadIdx.x >= 48) return;
  const int c = threadIdx.x;
  const float* invq = inv + b * C_DIM + h * HD;
  const float* invk = inv + 1536 + b * C_DIM + h * HD;
  const float scq = invq[c] * temp[h];
  const float* lp = logits + ((size_t)bh * 48 + c) * 48;
  float v[48];
  float mx = -1e30f;
#pragma unroll
  for (int d = 0; d < 48; ++d) {
    v[d] = lp[d] * scq * invk[d];
    mx = fmaxf(mx, v[d]);
  }
  float s = 0.f;
#pragma unroll
  for (int d = 0; d < 48; ++d) {
    v[d] = __expf(v[d] - mx);
    s += v[d];
  }
  const float rs = 1.f / s;
  float* op = attn_out + ((size_t)bh * 48 + c) * 48;
#pragma unroll
  for (int d = 0; d < 48; ++d) op[d] = v[d] * rs;
}

// ---------------------------------------------------------------------------
// K5: av_t[b, p, c'] = sum_d attn[b,h,c,d] * v[b,h,d,p]  (K=48 padded to 64)
// attn tile staged into LDS by the Tensor Data Mover (TDM).
// Output stored TRANSPOSED: av_t[(b*N_PIX + p)*C_DIM + h*48 + c] (bf16).
// grid (1024, 32), block 96 (3 waves).
// ---------------------------------------------------------------------------
__global__ void av_gemm_kernel(const bf16_t* __restrict__ qkv2,
                               const float* __restrict__ attn,
                               bf16_t* __restrict__ av_t) {
  const int bh = blockIdx.y, b = bh >> 2, h = bh & 3;
  const int lane = threadIdx.x & 31, wave = threadIdx.x >> 5;
  const int m0 = wave * 16, n0 = blockIdx.x * 16;
  const int half = lane >> 4, ln = lane & 15;

  __shared__ float atile[48 * 48];

  // --- TDM: DMA the 48x48 fp32 attn tile for this (b,h) into LDS ----------
  if (threadIdx.x < 32) {                          // one wave issues (uniform)
    const uint64_t ga = (uint64_t)(uintptr_t)(attn + (size_t)bh * 48 * 48);
    const uint32_t lds_addr = (uint32_t)(uintptr_t)atile;
    u32x4 g0;
    g0[0] = 1u;                                    // count=1, user descriptor
    g0[1] = lds_addr;                              // lds_addr (bytes)
    g0[2] = (uint32_t)ga;                          // global_addr[31:0]
    g0[3] = (uint32_t)((ga >> 32) & 0x1FFFFFFu) | (2u << 30);  // [56:32]|type=2
    i32x8 g1;
    g1[0] = (2 << 16);                             // data_size = 4B
    g1[1] = (48 << 16);                            // tensor_dim0 = 48
    g1[2] = (48 << 16);                            // tensor_dim1 = 48
    g1[3] = (48 << 16);                            // tile_dim0 = 48
    g1[4] = 48;                                    // tile_dim1 = 48, tile_dim2=0
    g1[5] = 48;                                    // tensor_dim0_stride = 48
    g1[6] = 0;
    g1[7] = 0;
    i32x4 gz = {0, 0, 0, 0};
    i32x8 gz8 = {0, 0, 0, 0, 0, 0, 0, 0};
    __builtin_amdgcn_tensor_load_to_lds(g0, g1, gz, gz, gz8, 0);
    __builtin_amdgcn_s_wait_tensorcnt(0);
  }
  __syncthreads();

  const bf16_t* vmat = qkv2 + ((size_t)(b * OC3 + 2 * C_DIM + h * HD) << 14);
  const bf16_t zero = (bf16_t)0.f;

  v8f acc = {};
#pragma unroll
  for (int k0 = 0; k0 < 64; k0 += 32) {
    v16bf a, bb;
#pragma unroll
    for (int i = 0; i < 16; ++i) {
      const int ka = k0 + (half << 4) + i;
      a[i] = (ka < HD) ? f2bf(atile[(m0 + ln) * 48 + ka]) : zero;  // ds loads
      const int kb = k0 + (half << 4) + i;
      bb[i] = (kb < HD) ? vmat[((size_t)kb << 14) + n0 + ln] : zero;
    }
    acc = wmma_bf16(a, bb, acc);
  }
  // Transposed store: one contiguous v8bf (16B) per lane.
  v8bf o;
#pragma unroll
  for (int r = 0; r < 8; ++r) o[r] = f2bf(acc[r]);
  bf16_t* op = av_t + ((size_t)(b * N_PIX + n0 + ln) * C_DIM) + h * HD + m0 + (half << 3);
  *(v8bf*)op = o;
}

// ---------------------------------------------------------------------------
// K6: out[b,o,p] = sum_c w_proj[o,c] * av_t[b,p,c]   (192x16384, K=192), f32
// grid (1024, 3, 8), block 128. B fragment = one 32B load (av is transposed).
// ---------------------------------------------------------------------------
__global__ void proj_gemm_kernel(const bf16_t* __restrict__ av_t,
                                 const float* __restrict__ w,
                                 float* __restrict__ out) {
  const int lane = threadIdx.x & 31, wave = threadIdx.x >> 5;
  const int half = lane >> 4, ln = lane & 15;
  const int n0 = blockIdx.x * 16;
  const int m0 = (blockIdx.y * 4 + wave) * 16;
  const int b  = blockIdx.z;
  const float*  wrow = w + (size_t)(m0 + ln) * C_DIM;
  const bf16_t* brow = av_t + (size_t)(b * N_PIX + n0 + ln) * C_DIM;

  v8f acc = {};
  for (int k0 = 0; k0 < C_DIM; k0 += 32) {
    v16bf a  = load_a_f32(wrow, k0, half);
    v16bf bb = load_b_bf16(brow, k0, half);
    acc = wmma_bf16(a, bb, acc);
  }
  float* op = out + ((size_t)(b * C_DIM) << 14);
#pragma unroll
  for (int r = 0; r < 8; ++r)
    op[((size_t)(m0 + r + (half << 3)) << 14) + n0 + ln] = acc[r];
}

// ---------------------------------------------------------------------------
// Launch
// ---------------------------------------------------------------------------
extern "C" void kernel_launch(void* const* d_in, const int* in_sizes, int n_in,
                              void* d_out, int out_size, void* d_ws, size_t ws_size,
                              hipStream_t stream) {
  const float* x      = (const float*)d_in[0];
  const float* w_qkv  = (const float*)d_in[1];
  const float* w_dw   = (const float*)d_in[2];
  const float* w_proj = (const float*)d_in[3];
  const float* temp   = (const float*)d_in[4];

  float* out      = (float*)d_out;
  float* attn_out = out + (size_t)B_DIM * C_DIM * N_PIX;

  // Workspace:
  //  [0, 151MB)      qkv_raw (bf16)  -> recycled after K2:
  //                    invn   @ 0        (12 KB)
  //                    logits @ 64 KB    (288 KB, fp32)
  //                    av_t   @ 1 MB     (48 MB, bf16, transposed)
  //  [151MB, 302MB)  qkv_conv (bf16)
  char* ws = (char*)d_ws;
  const size_t qkv_elems = (size_t)B_DIM * OC3 * N_PIX;   // 75,497,472
  bf16_t* qkv_raw  = (bf16_t*)ws;
  bf16_t* qkv_conv = (bf16_t*)(ws + qkv_elems * sizeof(bf16_t));
  float*  invn     = (float*)ws;
  float*  logits   = (float*)(ws + (64 << 10));
  bf16_t* av_t     = (bf16_t*)(ws + (1 << 20));

  {
    dim3 g(N_PIX / 16, OC3 / 64, B_DIM);                  // (1024, 9, 8)
    qkv_gemm_kernel<<<g, 128, 0, stream>>>(x, w_qkv, qkv_raw);
  }
  dwconv_kernel<<<(unsigned)(qkv_elems / 256), 256, 0, stream>>>(qkv_raw, w_dw,
                                                                 qkv_conv);
  rownorm_kernel<<<2 * B_DIM * C_DIM, 256, 0, stream>>>(qkv_conv, invn);

  (void)hipMemsetAsync(logits, 0, (size_t)32 * 48 * 48 * sizeof(float), stream);
  {
    dim3 g(NCHUNK, B_DIM * HEADS);                        // (8, 32)
    attn_partial_kernel<<<g, 288, 0, stream>>>(qkv_conv, logits);
  }
  attn_softmax_kernel<<<B_DIM * HEADS, 64, 0, stream>>>(logits, invn, temp,
                                                        attn_out);
  {
    dim3 g(N_PIX / 16, B_DIM * HEADS);                    // (1024, 32)
    av_gemm_kernel<<<g, 96, 0, stream>>>(qkv_conv, attn_out, av_t);
  }
  {
    dim3 g(N_PIX / 16, C_DIM / 64, B_DIM);                // (1024, 3, 8)
    proj_gemm_kernel<<<g, 128, 0, stream>>>(av_t, w_proj, out);
  }
}
